// Flashback_87230785782295
// MI455X (gfx1250) — matrix operand order repaired and checked by reading, not compiled, not run on hardware
//
#include <hip/hip_runtime.h>
#include <hip/hip_bf16.h>
#include <math.h>

typedef __bf16 bf16;
typedef __attribute__((ext_vector_type(8)))  __bf16 v8bf;
typedef __attribute__((ext_vector_type(16))) __bf16 v16bf;
typedef __attribute__((ext_vector_type(8)))  float  v8f;

constexpr int kSeq  = 128;
constexpr int kUB   = 256;
constexpr int kH    = 128;
constexpr int kH2   = 256;
constexpr int kNloc = 8000;
constexpr int kDeg  = 16;
constexpr long long kM = (long long)kSeq * kUB;   // 32768 rows (s-major, u-minor)

// ---------------- WMMA bf16 helpers (16x16x32, f32 accum) ----------------

__device__ __forceinline__ v16bf bfcat(v8bf lo, v8bf hi) {
  return __builtin_shufflevector(lo, hi, 0,1,2,3,4,5,6,7,8,9,10,11,12,13,14,15);
}

__device__ __forceinline__ v8f wmma_bf16(v16bf a, v16bf b, v8f c) {
  return __builtin_amdgcn_wmma_f32_16x16x32_bf16(false, a, false, b, (short)0, c,
                                                 false, false);
}

// A fragment: rowA = pointer to this lane's M row (row-major, contiguous K).
// ISA layout: lanes 0-15 hold K = k0+0..7 & k0+16..23 ; lanes 16-31 hold K = k0+8..15 & k0+24..31.
__device__ __forceinline__ v16bf load_afrag(const bf16* rowA, int k0, int half) {
  const bf16* p = rowA + k0 + half * 8;
  return bfcat(*(const v8bf*)p, *(const v8bf*)(p + 16));
}

// B fragment: rowB = pointer to this lane's N column of B stored transposed [N][K].
// ISA layout: lanes 0-15 hold K = k0+0..7 & k0+16..23 ; lanes 16-31 hold K = k0+8..15 & k0+24..31.
__device__ __forceinline__ v16bf load_bfrag(const bf16* rowB, int k0, int half) {
  const bf16* p = rowB + k0 + half * 8;
  return bfcat(*(const v8bf*)p, *(const v8bf*)(p + 16));
}

__device__ __forceinline__ v8f gemm_tile(const bf16* rowA, const bf16* rowB,
                                         int K, int half) {
  v8f c = {};
  for (int k0 = 0; k0 < K; k0 += 32)
    c = wmma_bf16(load_afrag(rowA, k0, half), load_bfrag(rowB, k0, half), c);
  return c;
}
// C layout: VGPR g -> row = g + 8*half (lanes<16 half=0), col = n0 + (lane&15).

// ---------------- prep kernels ----------------

// dst[N][K] bf16 = transpose of src[K][N] f32 (write-coalesced)
__global__ void k_transpose_bf16(bf16* __restrict__ dst, const float* __restrict__ src,
                                 int K, int N) {
  int idx = blockIdx.x * blockDim.x + threadIdx.x;
  if (idx >= K * N) return;
  int n = idx / K, k = idx % K;
  dst[idx] = (bf16)src[(long long)k * N + n];
}

// graph aggregate: row i owns edges [i*16, i*16+16) plus self-loop at 128000+i
__global__ void k_gconv_agg(bf16* __restrict__ agg, const int* __restrict__ cols,
                            const float* __restrict__ vals,
                            const float* __restrict__ loc_emb) {
  int i = blockIdx.x, hh = threadIdx.x;          // blockDim = 128
  float acc = 0.f;
  #pragma unroll
  for (int e = 0; e < kDeg; ++e) {
    int idx = i * kDeg + e;
    acc += vals[idx] * loc_emb[(long long)cols[idx] * kH + hh];
  }
  int sidx = kNloc * kDeg + i;
  acc += vals[sidx] * loc_emb[(long long)cols[sidx] * kH + hh];
  agg[(long long)i * kH + hh] = (bf16)acc;
}

__global__ void k_gather_pu(bf16* __restrict__ pu, const float* __restrict__ user_emb,
                            const int* __restrict__ active_user) {
  int idx = blockIdx.x * blockDim.x + threadIdx.x;   // < 256*128
  int u = idx >> 7, k = idx & 127;
  pu[idx] = (bf16)user_emb[(long long)active_user[u] * kH + k];
}

// ---------------- GEMM kernels ----------------

// encW = agg @ W_gconv + b_gconv  -> bf16 [8000][128]
__global__ void k_gemm_encw(bf16* __restrict__ encw, const bf16* __restrict__ agg,
                            const bf16* __restrict__ wgt, const float* __restrict__ bg) {
  int lane = threadIdx.x & 31, wv = threadIdx.x >> 5, half = lane >> 4, r = lane & 15;
  int m0 = blockIdx.x * 16, n0 = wv * 16;
  v8f c = gemm_tile(agg + (long long)(m0 + r) * kH, wgt + (long long)(n0 + r) * kH,
                    kH, half);
  int n = n0 + r; float b = bg[n];
  #pragma unroll
  for (int g = 0; g < 8; ++g)
    encw[(long long)(m0 + g + 8 * half) * kH + n] = (bf16)(c[g] + b);
}

// p_proj = tanh(p_u @ W_proj + b_proj)  f32 [256][256]
__global__ void k_gemm_pproj(float* __restrict__ pproj, const bf16* __restrict__ pu,
                             const bf16* __restrict__ wpt, const float* __restrict__ bp) {
  int lane = threadIdx.x & 31, wv = threadIdx.x >> 5, half = lane >> 4, r = lane & 15;
  int m0 = blockIdx.x * 16;
  int n0 = (blockIdx.y * 8 + wv) * 16;
  v8f c = gemm_tile(pu + (long long)(m0 + r) * kH, wpt + (long long)(n0 + r) * kH,
                    kH, half);
  int n = n0 + r; float b = bp[n];
  #pragma unroll
  for (int g = 0; g < 8; ++g)
    pproj[(long long)(m0 + g + 8 * half) * kH2 + n] = tanhf(c[g] + b);
}

// xih[m][n] = x_emb[m] @ W_ih + b_ih + b_hh  (A rows gathered via x[m])
__global__ void k_gemm_xih(float* __restrict__ xih, const bf16* __restrict__ encw,
                           const int* __restrict__ x, const bf16* __restrict__ wiht,
                           const float* __restrict__ bih, const float* __restrict__ bhh) {
  int lane = threadIdx.x & 31, wv = threadIdx.x >> 5, half = lane >> 4, r = lane & 15;
  int m0 = blockIdx.x * 16, n0 = wv * 16;
  const bf16* rowA = encw + (long long)x[m0 + r] * kH;
  v8f c = gemm_tile(rowA, wiht + (long long)(n0 + r) * kH, kH, half);
  int n = n0 + r; float b = bih[n] + bhh[n];
  #pragma unroll
  for (int g = 0; g < 8; ++g)
    xih[(long long)(m0 + g + 8 * half) * kH + n] = c[g] + b;
}

// sim_acc[m] = sum_n (p_proj[u][n] + pref[n] - tanh(x_emb[m]@W_proj + b)[n])^2
// one block per M-tile; 8 waves x 2 N-tiles = full N=256; deterministic LDS reduce
__global__ void k_sim_acc(float* __restrict__ simacc, const bf16* __restrict__ encw,
                          const int* __restrict__ x, const bf16* __restrict__ wpt,
                          const float* __restrict__ bp, const float* __restrict__ pproj,
                          const float* __restrict__ pref) {
  __shared__ float part[8][16];
  int lane = threadIdx.x & 31, wv = threadIdx.x >> 5, half = lane >> 4, r = lane & 15;
  int m0 = blockIdx.x * 16;
  const bf16* rowA = encw + (long long)x[m0 + r] * kH;
  float rowacc[8] = {0.f,0.f,0.f,0.f,0.f,0.f,0.f,0.f};
  #pragma unroll
  for (int tt = 0; tt < 2; ++tt) {
    int n0 = (wv + tt * 8) * 16;
    v8f c = gemm_tile(rowA, wpt + (long long)(n0 + r) * kH, kH, half);
    int n = n0 + r;
    float b = bp[n], pe = pref[n];
    #pragma unroll
    for (int g = 0; g < 8; ++g) {
      int m = m0 + g + 8 * half;
      int u = m & (kUB - 1);
      float xp = tanhf(c[g] + b);
      float d = pproj[(long long)u * kH2 + n] + pe - xp;
      rowacc[g] += d * d;
    }
  }
  #pragma unroll
  for (int g = 0; g < 8; ++g) {
    float v = rowacc[g];
    for (int off = 8; off; off >>= 1) v += __shfl_xor(v, off, 32);
    if (r == 0) part[wv][g + 8 * half] = v;
  }
  __syncthreads();
  if (threadIdx.x < 16) {
    float sum = 0.f;
    #pragma unroll
    for (int w = 0; w < 8; ++w) sum += part[w][threadIdx.x];
    simacc[m0 + threadIdx.x] = sum;
  }
}

__global__ void k_sim_fin(float* __restrict__ sim) {
  int m = blockIdx.x * blockDim.x + threadIdx.x;
  if (m < (int)kM) sim[m] = 1.f / (1.f + expf(sqrtf(sim[m] + 1e-12f)));
}

// wts[u][i][j] bf16 (tril-masked) and sum_w[i*256+u]; one wave per (u,i)
__global__ void k_wts(bf16* __restrict__ wts, float* __restrict__ sumw,
                      const float* __restrict__ sim, const int* __restrict__ t,
                      const int* __restrict__ s) {
  int lane = threadIdx.x & 31;
  int gw = blockIdx.x * 8 + (threadIdx.x >> 5);
  int u = gw >> 7;          // / 128
  int i = gw & 127;
  float ti  = (float)t[i * kUB + u];
  float si0 = (float)s[(i * kUB + u) * 2 + 0];
  float si1 = (float)s[(i * kUB + u) * 2 + 1];
  const float c1 = 6.28318530717958647692f / 86400.0f;
  const float c2 = -0.1f / 86400.0f;
  float lsum = 0.f;
  bf16* wrow = wts + ((long long)u * kSeq + i) * kSeq;
  #pragma unroll
  for (int jj = 0; jj < 4; ++jj) {
    int j = lane + jj * 32;
    float w = 0.f;
    if (j <= i) {
      float dt = ti - (float)t[j * kUB + u];
      float a = (cosf(dt * c1) + 1.f) * 0.5f * expf(dt * c2);
      float d0 = si0 - (float)s[(j * kUB + u) * 2 + 0];
      float d1 = si1 - (float)s[(j * kUB + u) * 2 + 1];
      float b = expf(-sqrtf(d0 * d0 + d1 * d1) * 0.1f);
      w = (a * b + 1e-10f) * sim[j * kUB + u];
    }
    wrow[j] = (bf16)w;
    lsum += w;
  }
  for (int off = 16; off; off >>= 1) lsum += __shfl_xor(lsum, off, 32);
  if (lane == 0) sumw[i * kUB + u] = lsum;
}

// RNN: 16 blocks x 16 users; h (bf16) and W_hh^T staged in LDS, rows padded to 136
__global__ void k_rnn(float* __restrict__ out, const float* __restrict__ xih,
                      const bf16* __restrict__ whht, const float* __restrict__ h0) {
  __shared__ __align__(16) bf16 whh[128 * 136];
  __shared__ __align__(16) bf16 hls[16 * 136];
  int tid = threadIdx.x;
  int lane = tid & 31, wv = tid >> 5, half = lane >> 4, r = lane & 15;
  int u0 = blockIdx.x * 16;
  for (int idx = tid; idx < 128 * 128; idx += 256)
    whh[(idx >> 7) * 136 + (idx & 127)] = whht[idx];
  for (int idx = tid; idx < 16 * 128; idx += 256)
    hls[(idx >> 7) * 136 + (idx & 127)] =
        (bf16)h0[(long long)(u0 + (idx >> 7)) * kH + (idx & 127)];
  __syncthreads();
  int n0 = wv * 16, n = n0 + r;
  const bf16* rowA = hls + r * 136;
  const bf16* rowB = whh + (n0 + r) * 136;
  for (int st = 0; st < kSeq; ++st) {
    v8f c = gemm_tile(rowA, rowB, kH, half);
    float vals[8];
    #pragma unroll
    for (int g = 0; g < 8; ++g) {
      int u = u0 + g + 8 * half;
      vals[g] = tanhf(c[g] + xih[((long long)st * kUB + u) * kH + n]);
    }
    __syncthreads();   // all reads of hls done
    #pragma unroll
    for (int g = 0; g < 8; ++g) {
      int ul = g + 8 * half;
      hls[ul * 136 + n] = (bf16)vals[g];
      out[((long long)st * kUB + u0 + ul) * kH + n] = vals[g];
    }
    __syncthreads();   // new h visible
  }
}

// per-user einsum: A_fc[i*256+u][h] = (wts[u] @ out[:,u,:]) / sum_w  (bf16)
__global__ void k_outw(bf16* __restrict__ afc, const float* __restrict__ out,
                       const bf16* __restrict__ wts, const float* __restrict__ sumw) {
  __shared__ __align__(16) bf16 outT[128 * 136];     // [h][j]
  int u = blockIdx.x;
  int tid = threadIdx.x, lane = tid & 31, wv = tid >> 5, half = lane >> 4, r = lane & 15;
  for (int idx = tid; idx < 128 * 128; idx += 256) {
    int j = idx >> 7, hc = idx & 127;
    outT[hc * 136 + j] = (bf16)out[((long long)j * kUB + u) * kH + hc];
  }
  __syncthreads();
  const bf16* wbase = wts + (long long)u * kSeq * kSeq;
  for (int tt = wv; tt < 64; tt += 8) {
    int it = tt >> 3, ht = tt & 7;
    int i0 = it * 16, h0 = ht * 16;
    v8f c = gemm_tile(wbase + (long long)(i0 + r) * kSeq, outT + (h0 + r) * 136,
                      kSeq, half);
    int hcol = h0 + r;
    #pragma unroll
    for (int g = 0; g < 8; ++g) {
      int i = i0 + g + 8 * half;
      int m = i * kUB + u;
      afc[(long long)m * kH + hcol] = (bf16)(c[g] / sumw[m]);
    }
  }
}

// upre[u][n] = p_u @ W_fc[128:256,:] + b_fc   (constant across seq)
__global__ void k_upre(float* __restrict__ upre, const bf16* __restrict__ pu,
                       const bf16* __restrict__ wfct, const float* __restrict__ bfc) {
  int lane = threadIdx.x & 31, wv = threadIdx.x >> 5, half = lane >> 4, r = lane & 15;
  int n0 = blockIdx.x * 16;
  int m0 = (blockIdx.y * 8 + wv) * 16;
  v8f c = gemm_tile(pu + (long long)(m0 + r) * kH,
                    wfct + (long long)(n0 + r) * kH2 + kH, kH, half);
  int n = n0 + r; float b = bfc[n];
  #pragma unroll
  for (int g = 0; g < 8; ++g)
    upre[(long long)(m0 + g + 8 * half) * kNloc + n] = c[g] + b;
}

// y[m][n] = A_fc[m] @ W_fc[0:128,:] + upre[u][n]; each wave: 1 M-tile x 4 N-tiles
__global__ void k_fc(float* __restrict__ y, const bf16* __restrict__ afc,
                     const bf16* __restrict__ wfct, const float* __restrict__ upre) {
  int lane = threadIdx.x & 31, wv = threadIdx.x >> 5, half = lane >> 4, r = lane & 15;
  int n0 = blockIdx.x * 64;
  int m0 = (blockIdx.y * 8 + wv) * 16;
  const bf16* rowA = afc + (long long)(m0 + r) * kH;
  v16bf af[4];
  #pragma unroll
  for (int kk = 0; kk < 4; ++kk) af[kk] = load_afrag(rowA, kk * 32, half);
  int ub0 = m0 & (kUB - 1);
  #pragma unroll
  for (int nt = 0; nt < 4; ++nt) {
    const bf16* rowB = wfct + (long long)(n0 + nt * 16 + r) * kH2;
    v8f c = {};
    #pragma unroll
    for (int kk = 0; kk < 4; ++kk)
      c = wmma_bf16(af[kk], load_bfrag(rowB, kk * 32, half), c);
    int n = n0 + nt * 16 + r;
    #pragma unroll
    for (int g = 0; g < 8; ++g) {
      int row = g + 8 * half;
      y[(long long)(m0 + row) * kNloc + n] = c[g] + upre[(long long)(ub0 + row) * kNloc + n];
    }
  }
}

__global__ void k_hlast(float* __restrict__ dst, const float* __restrict__ out) {
  int idx = blockIdx.x * blockDim.x + threadIdx.x;
  if (idx < kUB * kH) dst[idx] = out[(long long)(kSeq - 1) * kUB * kH + idx];
}

// ---------------- launcher ----------------

extern "C" void kernel_launch(void* const* d_in, const int* in_sizes, int n_in,
                              void* d_out, int out_size, void* d_ws, size_t ws_size,
                              hipStream_t stream) {
  (void)in_sizes; (void)n_in; (void)out_size; (void)ws_size;
  const int*   x        = (const int*)d_in[0];
  const int*   t        = (const int*)d_in[1];
  const int*   sxy      = (const int*)d_in[3];
  const float* h0       = (const float*)d_in[7];
  const int*   act_user = (const int*)d_in[8];
  const int*   ecols    = (const int*)d_in[10];
  const float* evals    = (const float*)d_in[11];
  const float* loc_emb  = (const float*)d_in[12];
  const float* user_emb = (const float*)d_in[13];
  const float* pref     = (const float*)d_in[14];
  const float* W_proj   = (const float*)d_in[15];
  const float* b_proj   = (const float*)d_in[16];
  const float* W_gconv  = (const float*)d_in[17];
  const float* b_gconv  = (const float*)d_in[18];
  const float* W_ih     = (const float*)d_in[19];
  const float* W_hh     = (const float*)d_in[20];
  const float* b_ih     = (const float*)d_in[21];
  const float* b_hh     = (const float*)d_in[22];
  const float* W_fc     = (const float*)d_in[23];
  const float* b_fc     = (const float*)d_in[24];
  float* y = (float*)d_out;

  char* ws = (char*)d_ws;
  size_t o = 0;
  auto alloc = [&](size_t bytes) { size_t p = o; o += (bytes + 255) & ~(size_t)255; return p; };
  bf16*  wgt_t   = (bf16*)(ws + alloc((size_t)kH * kH * 2));
  bf16*  wih_t   = (bf16*)(ws + alloc((size_t)kH * kH * 2));
  bf16*  whh_t   = (bf16*)(ws + alloc((size_t)kH * kH * 2));
  bf16*  wproj_t = (bf16*)(ws + alloc((size_t)kH2 * kH * 2));
  bf16*  wfc_t   = (bf16*)(ws + alloc((size_t)kNloc * kH2 * 2));
  bf16*  agg     = (bf16*)(ws + alloc((size_t)kNloc * kH * 2));
  bf16*  encw    = (bf16*)(ws + alloc((size_t)kNloc * kH * 2));
  bf16*  pu      = (bf16*)(ws + alloc((size_t)kUB * kH * 2));
  float* pproj   = (float*)(ws + alloc((size_t)kUB * kH2 * 4));
  float* xih     = (float*)(ws + alloc((size_t)kM * kH * 4));
  float* outb    = (float*)(ws + alloc((size_t)kM * kH * 4));
  float* simac   = (float*)(ws + alloc((size_t)kM * 4));
  bf16*  wtsb    = (bf16*)(ws + alloc((size_t)kUB * kSeq * kSeq * 2));
  float* sumw    = (float*)(ws + alloc((size_t)kM * 4));
  bf16*  afc     = (bf16*)(ws + alloc((size_t)kM * kH * 2));
  float* upre    = (float*)(ws + alloc((size_t)kUB * kNloc * 4));

  // weight transposes (f32 -> bf16 [N][K])
  k_transpose_bf16<<<(kH * kH + 255) / 256, 256, 0, stream>>>(wgt_t, W_gconv, kH, kH);
  k_transpose_bf16<<<(kH * kH + 255) / 256, 256, 0, stream>>>(wih_t, W_ih, kH, kH);
  k_transpose_bf16<<<(kH * kH + 255) / 256, 256, 0, stream>>>(whh_t, W_hh, kH, kH);
  k_transpose_bf16<<<(kH * kH2 + 255) / 256, 256, 0, stream>>>(wproj_t, W_proj, kH, kH2);
  k_transpose_bf16<<<(kH2 * kNloc + 255) / 256, 256, 0, stream>>>(wfc_t, W_fc, kH2, kNloc);

  // gathers + graph aggregate
  k_gather_pu<<<(kUB * kH + 255) / 256, 256, 0, stream>>>(pu, user_emb, act_user);
  k_gconv_agg<<<kNloc, kH, 0, stream>>>(agg, ecols, evals, loc_emb);

  // encoder_weight, p_proj, xih, sim
  k_gemm_encw<<<kNloc / 16, 256, 0, stream>>>(encw, agg, wgt_t, b_gconv);
  k_gemm_pproj<<<dim3(kUB / 16, 2), 256, 0, stream>>>(pproj, pu, wproj_t, b_proj);
  k_gemm_xih<<<(int)(kM / 16), 256, 0, stream>>>(xih, encw, x, wih_t, b_ih, b_hh);
  k_sim_acc<<<(int)(kM / 16), 256, 0, stream>>>(simac, encw, x, wproj_t, b_proj, pproj, pref);
  k_sim_fin<<<(int)((kM + 255) / 256), 256, 0, stream>>>(simac);

  // spatiotemporal weights + RNN + per-user einsum
  k_wts<<<(int)(kM / 8), 256, 0, stream>>>(wtsb, sumw, simac, t, sxy);
  k_rnn<<<kUB / 16, 256, 0, stream>>>(outb, xih, whh_t, h0);
  k_outw<<<kUB, 256, 0, stream>>>(afc, outb, wtsb, sumw);

  // final projection: user-constant half once, then big store-bound GEMM
  k_upre<<<dim3(kNloc / 16, 2), 256, 0, stream>>>(upre, pu, wfc_t, b_fc);
  k_fc<<<dim3(kNloc / 64, (int)(kM / 128)), 256, 0, stream>>>(y, afc, wfc_t, upre);

  // h_last appended after y_linear
  k_hlast<<<(kUB * kH + 255) / 256, 256, 0, stream>>>(y + kM * kNloc, outb);
}
